// OptimalMultiHeadAttention_49950469653279
// MI455X (gfx1250) — matrix operand order repaired and checked by reading, not compiled, not run on hardware
//
#include <hip/hip_runtime.h>
#include <hip/hip_bf16.h>

// ---------------------------------------------------------------------------
// B=2, S=2048, HID=2048, 32 heads (8 KV x 4 groups), HEAD_DIM=64, WINDOW=128.
// d_out = [output (2*2048*2048 f32)] ++ [attn_weights (2*32*2048*2048 f32)]
// Pipeline: f32->bf16 once, bf16 WMMA everywhere, f32 accum/softmax.
// ---------------------------------------------------------------------------
#define SB     2
#define SLEN   2048
#define HID    2048
#define NHEADS 32
#define NKV    8
#define HDIM   64
#define WIN    128

typedef __attribute__((ext_vector_type(16))) __bf16        v16bf;
typedef __attribute__((ext_vector_type(8)))  float         v8f;
typedef __attribute__((ext_vector_type(8)))  unsigned int  v8u;
typedef unsigned short ushort_t;

#define WMMA_BF16(A, Bm, C) \
  __builtin_amdgcn_wmma_f32_16x16x32_bf16(false, (A), false, (Bm), (short)0, (C), false, false)

// ---------------- async global<->LDS (ASYNCcnt) ----------------------------
static __device__ __forceinline__ unsigned lds_off(const void* p) {
  return (unsigned)(unsigned long long)p;     // ISA 10.2: LDS address is the low 32 bits of the pointer
}
static __device__ __forceinline__ void async_load_b128(unsigned dst_lds, const void* gsrc) {
  asm volatile("global_load_async_to_lds_b128 %0, %1, off"
               :: "v"(dst_lds), "v"(gsrc) : "memory");
}
static __device__ __forceinline__ void async_store_b128(void* gdst, unsigned src_lds) {
  asm volatile("global_store_async_from_lds_b128 %0, %1, off"
               :: "v"(gdst), "v"(src_lds) : "memory");
}
#define WAIT_ASYNC(n) asm volatile("s_wait_asynccnt %0" :: "n"(n) : "memory")

// ---------------- bf16 conversion (RNE) -------------------------------------
static __device__ __forceinline__ ushort_t f2bf(float f) {
  unsigned u = __builtin_bit_cast(unsigned, f);
  u += 0x7FFFu + ((u >> 16) & 1u);
  return (ushort_t)(u >> 16);
}
static __device__ __forceinline__ unsigned pk2bf(float lo, float hi) {
  return (unsigned)f2bf(lo) | ((unsigned)f2bf(hi) << 16);
}

// ---------------- WMMA fragment gathers (ISA 7.12.2, wave32) ----------------
// A 16x32 bf16: lane L: M=L%16, half=L/16; 16 contiguous bf16 at K=half*16.
static __device__ __forceinline__ v16bf load_frag_a_bf(const ushort_t* lds, int m,
                                                       int kbase, int stride, int half) {
  const uint4 a = *(const uint4*)(lds + m * stride + kbase + half * 16);
  const uint4 b = *(const uint4*)(lds + m * stride + kbase + half * 16 + 8);
  v8u r;
  r[0] = a.x; r[1] = a.y; r[2] = a.z; r[3] = a.w;
  r[4] = b.x; r[5] = b.y; r[6] = b.z; r[7] = b.w;
  return __builtin_bit_cast(v16bf, r);
}
// B 32x16 bf16 from [n][k] LDS: lane L: N=L%16; 16 contiguous bf16 at K=half*16.
static __device__ __forceinline__ v16bf load_frag_b_bf(const ushort_t* lds, int n,
                                                       int kbase, int stride, int half) {
  const uint4* p = (const uint4*)(lds + n * stride + kbase + half * 16);
  const uint4 a = p[0], b = p[1];
  v8u r;
  r[0] = a.x; r[1] = a.y; r[2] = a.z; r[3] = a.w;
  r[4] = b.x; r[5] = b.y; r[6] = b.z; r[7] = b.w;
  return __builtin_bit_cast(v16bf, r);
}
// B 32x16 bf16 from k-major [k][n] LDS (V tile): strided u16 gather.
static __device__ __forceinline__ v16bf load_frag_b_u16T(const ushort_t* lds, int n,
                                                         int stride, int half) {
  v8u r;
#pragma unroll
  for (int d = 0; d < 8; ++d) {
    unsigned lo = lds[(half * 16 + 2 * d) * stride + n];
    unsigned hi = lds[(half * 16 + 2 * d + 1) * stride + n];
    r[d] = lo | (hi << 16);
  }
  return __builtin_bit_cast(v16bf, r);
}
// A 16x32 from f32 LDS (probabilities): convert on gather.
static __device__ __forceinline__ v16bf load_frag_a_f32(const float* lds, int m,
                                                        int kbase, int stride, int half) {
  v8u r;
#pragma unroll
  for (int d = 0; d < 8; ++d) {
    int koff = half * 16 + 2 * d;
    r[d] = pk2bf(lds[m * stride + kbase + koff], lds[m * stride + kbase + koff + 1]);
  }
  return __builtin_bit_cast(v16bf, r);
}

// ---------------------------------------------------------------------------
// One-time f32 -> bf16 conversion (bandwidth-bound, vectorized x4).
// ---------------------------------------------------------------------------
__global__ void __launch_bounds__(256)
cvt_f32_bf16(const float* __restrict__ src, ushort_t* __restrict__ dst, int n4) {
  int i = blockIdx.x * 256 + threadIdx.x;
  if (i < n4) {
    float4 v = ((const float4*)src)[i];
    uint2 o;
    o.x = pk2bf(v.x, v.y);
    o.y = pk2bf(v.z, v.w);
    ((uint2*)dst)[i] = o;
  }
}

// ---------------------------------------------------------------------------
// C[M,N] = A[M,K] @ W[N,K]^T, A/W bf16, async double-buffered staging.
// 128 threads, 64x64 macro tile, K-step 32. LDS row stride 40 bf16 (80B).
// BF16OUT: store C as bf16 (Q/K/V projections) or f32 (final O projection).
// ---------------------------------------------------------------------------
template <bool BF16OUT>
__global__ void __launch_bounds__(128)
gemm_xwt(const ushort_t* __restrict__ A, const ushort_t* __restrict__ W,
         void* __restrict__ C, int M, int N, int K) {
  __shared__ __align__(16) ushort_t Asf[2][64 * 40];
  __shared__ __align__(16) ushort_t Bsf[2][64 * 40];

  const int n0 = blockIdx.x * 64;
  const int m0 = blockIdx.y * 64;
  const int tid = threadIdx.x;
  const int lane = tid & 31;
  const int wave = tid >> 5;
  const int half = lane >> 4;
  const int l16  = lane & 15;

  v8f acc[4] = {v8f{}, v8f{}, v8f{}, v8f{}};
  const int T = K / 32;

  auto issue_tile = [&](int t, int buf) {
    const int k0 = t * 32;
#pragma unroll
    for (int c = 0; c < 2; ++c) {          // 256 b128 chunks per matrix / 128 thr
      int idx = tid + c * 128;
      int r = idx >> 2, k8 = (idx & 3) * 8;
      async_load_b128(lds_off(&Asf[buf][r * 40 + k8]), &A[(size_t)(m0 + r) * K + k0 + k8]);
      async_load_b128(lds_off(&Bsf[buf][r * 40 + k8]), &W[(size_t)(n0 + r) * K + k0 + k8]);
    }
  };

  issue_tile(0, 0);
  for (int t = 0; t < T; ++t) {
    const int buf = t & 1;
    if (t + 1 < T) {
      issue_tile(t + 1, buf ^ 1);
      WAIT_ASYNC(4);                       // tile t's 4 per-thread loads done
    } else {
      WAIT_ASYNC(0);
    }
    __syncthreads();

    v16bf a = load_frag_a_bf(Asf[buf], wave * 16 + l16, 0, 40, half);
#pragma unroll
    for (int q = 0; q < 4; ++q) {
      v16bf b = load_frag_b_bf(Bsf[buf], q * 16 + l16, 0, 40, half);
      acc[q] = WMMA_BF16(a, b, acc[q]);
    }
    __syncthreads();
  }

  // C/D layout: lane L: N=L%16; VGPR r: M = r + 8*(L/16)
#pragma unroll
  for (int q = 0; q < 4; ++q)
#pragma unroll
    for (int r = 0; r < 8; ++r) {
      int m = m0 + wave * 16 + r + 8 * half;
      int n = n0 + q * 16 + l16;
      if (BF16OUT) ((ushort_t*)C)[(size_t)m * N + n] = f2bf(acc[q][r]);
      else         ((float*)C)[(size_t)m * N + n]    = acc[q][r];
    }
}

// ---------------------------------------------------------------------------
// Banded GQA attention, one wave per (b, head, 16-query tile). bf16 Q/K/V.
// Keys in [q0-128, q0+143]: 17x16 tiles QK^T, 9x32 tiles P@V.
// ---------------------------------------------------------------------------
__global__ void __launch_bounds__(32)
attn_kernel(const ushort_t* __restrict__ Q, const ushort_t* __restrict__ Kb,
            const ushort_t* __restrict__ Vb, ushort_t* __restrict__ ctx,
            float* __restrict__ attn) {
  __shared__ __align__(16) ushort_t Qf[16 * 72];   // [q][d]
  __shared__ __align__(16) ushort_t Kf[16 * 72];   // [key][d]
  __shared__ __align__(16) ushort_t Vf[32 * 72];   // [key][d]
  __shared__ __align__(16) float    Ps[16 * 289];  // scores / probs f32
  __shared__ __align__(16) float    Arow[2][SLEN]; // attn row staging

  const int q0   = blockIdx.x * 16;
  const int bh   = blockIdx.y;                     // b*32 + head
  const int b    = bh >> 5;
  const int head = bh & 31;
  const int h    = head >> 2;                      // kv head
  const int lane = threadIdx.x;
  const int half = lane >> 4;
  const int l16  = lane & 15;
  const int jbase = q0 - WIN;

  const ushort_t* Qp = Q  + (size_t)b * SLEN * HID + (size_t)head * HDIM;
  const ushort_t* Kp = Kb + (size_t)b * SLEN * (NKV * HDIM) + (size_t)h * HDIM;
  const ushort_t* Vp = Vb + (size_t)b * SLEN * (NKV * HDIM) + (size_t)h * HDIM;

  // ---- stage 16x64 Q tile (async, 128 b128 chunks / 32 lanes) ----
#pragma unroll
  for (int c = 0; c < 4; ++c) {
    int idx = lane + c * 32;
    int r = idx >> 3, k8 = (idx & 7) * 8;
    async_load_b128(lds_off(&Qf[r * 72 + k8]), &Qp[(size_t)(q0 + r) * HID + k8]);
  }
  WAIT_ASYNC(0);
  __syncthreads();

  const v16bf qa0 = load_frag_a_bf(Qf, l16, 0, 72, half);
  const v16bf qa1 = load_frag_a_bf(Qf, l16, 32, 72, half);

  // ---- QK^T over 17 key tiles of 16 (OOB rows clamped; masked later) ----
  for (int t = 0; t < 17; ++t) {
    int jt = jbase + t * 16;
    __syncthreads();
#pragma unroll
    for (int c = 0; c < 4; ++c) {
      int idx = lane + c * 32;
      int r = idx >> 3, k8 = (idx & 7) * 8;
      int j = jt + r;
      j = j < 0 ? 0 : (j >= SLEN ? SLEN - 1 : j);
      async_load_b128(lds_off(&Kf[r * 72 + k8]), &Kp[(size_t)j * (NKV * HDIM) + k8]);
    }
    WAIT_ASYNC(0);
    __syncthreads();

    v8f sc = v8f{};
    v16bf b0 = load_frag_b_bf(Kf, l16, 0, 72, half);
    v16bf b1 = load_frag_b_bf(Kf, l16, 32, 72, half);
    sc = WMMA_BF16(qa0, b0, sc);
    sc = WMMA_BF16(qa1, b1, sc);
#pragma unroll
    for (int r = 0; r < 8; ++r)
      Ps[(r + 8 * half) * 289 + t * 16 + l16] = sc[r];
  }
  __syncthreads();

  // ---- masked softmax per row (lanes 0..15), scale 1/sqrt(64)=0.125 ----
  if (lane < 16) {
    const int i = q0 + lane;
    float mx = -3.0e38f;
    for (int c = 0; c < 272; ++c) {
      int j = jbase + c;
      bool ok = (j >= 0) && (j < SLEN) && (j >= i - WIN) && (j <= i + WIN);
      float s = ok ? Ps[lane * 289 + c] * 0.125f : -3.0e38f;
      mx = fmaxf(mx, s);
      Ps[lane * 289 + c] = s;
    }
    float sum = 0.f;
    for (int c = 0; c < 272; ++c) {
      float e = __expf(Ps[lane * 289 + c] - mx);
      sum += e;
      Ps[lane * 289 + c] = e;
    }
    float inv = 1.0f / sum;
    for (int c = 0; c < 272; ++c) Ps[lane * 289 + c] *= inv;
    for (int c = 272; c < 289; ++c) Ps[lane * 289 + c] = 0.f;  // pad for PV tile 9
  }
  __syncthreads();

  // ---- stream full attn rows: build in LDS, async-store b128 to global ----
  {
    float* arowg = attn + ((size_t)(b * NHEADS + head) * SLEN + q0) * SLEN;
    for (int r = 0; r < 16; ++r) {
      const int buf = r & 1;
      const int i = q0 + r;
      WAIT_ASYNC(16);                      // stores from row r-2 (same buf) done
#pragma unroll 4
      for (int c = 0; c < 16; ++c) {       // 512 float4 per row / 32 lanes
        int idx4 = lane + c * 32;
        float4 v;
        float* pv = (float*)&v;
#pragma unroll
        for (int u = 0; u < 4; ++u) {
          int j = idx4 * 4 + u;
          bool inband = (j >= i - WIN) && (j <= i + WIN);
          pv[u] = inband ? Ps[r * 289 + (j - jbase)] : 0.f;
        }
        ((float4*)Arow[buf])[idx4] = v;
      }
      __syncthreads();                     // LDS writes visible before async reads
#pragma unroll 4
      for (int c = 0; c < 16; ++c) {
        int idx4 = lane + c * 32;
        async_store_b128(&arowg[(size_t)r * SLEN + idx4 * 4],
                         lds_off(&Arow[buf][idx4 * 4]));
      }
    }
  }

  // ---- P @ V over 9 key tiles of 32 (V staged [key][dim] bf16) ----
  v8f oacc[4] = {v8f{}, v8f{}, v8f{}, v8f{}};
  for (int t = 0; t < 9; ++t) {
    int jt = jbase + t * 32;
    __syncthreads();
#pragma unroll
    for (int c = 0; c < 8; ++c) {          // 256 b128 chunks / 32 lanes
      int idx = lane + c * 32;
      int r = idx >> 3, k8 = (idx & 7) * 8;
      int j = jt + r;
      j = j < 0 ? 0 : (j >= SLEN ? SLEN - 1 : j);
      async_load_b128(lds_off(&Vf[r * 72 + k8]), &Vp[(size_t)j * (NKV * HDIM) + k8]);
    }
    WAIT_ASYNC(0);                         // also drains remaining row stores
    __syncthreads();

    v16bf pa = load_frag_a_f32(Ps, l16, t * 32, 289, half);
#pragma unroll
    for (int dt = 0; dt < 4; ++dt) {
      v16bf vb = load_frag_b_u16T(Vf, dt * 16 + l16, 72, half);
      oacc[dt] = WMMA_BF16(pa, vb, oacc[dt]);
    }
  }

  // ---- context writeback (bf16): ctx[b, q, head*64 + d] ----
  ushort_t* cp = ctx + ((size_t)b * SLEN + q0) * HID + (size_t)head * HDIM;
#pragma unroll
  for (int dt = 0; dt < 4; ++dt)
#pragma unroll
    for (int r = 0; r < 8; ++r) {
      int m = r + 8 * half;
      cp[(size_t)m * HID + dt * 16 + l16] = f2bf(oacc[dt][r]);
    }
}

// ---------------------------------------------------------------------------
extern "C" void kernel_launch(void* const* d_in, const int* in_sizes, int n_in,
                              void* d_out, int out_size, void* d_ws, size_t ws_size,
                              hipStream_t stream) {
  const float* query = (const float*)d_in[0];
  const float* key   = (const float*)d_in[1];
  const float* value = (const float*)d_in[2];
  const float* W_q   = (const float*)d_in[3];
  const float* W_k   = (const float*)d_in[4];
  const float* W_v   = (const float*)d_in[5];
  const float* W_o   = (const float*)d_in[6];

  float* out  = (float*)d_out;                        // [B,S,HID]
  float* attn = out + (size_t)SB * SLEN * HID;        // [B,32,S,S]

  const size_t M   = (size_t)SB * SLEN;               // 4096
  const size_t XE  = M * HID;                         // 8,388,608 (x / q / ctx)
  const size_t KVE = M * (NKV * HDIM);                // 2,097,152
  const size_t WQE = (size_t)HID * HID;               // 4,194,304
  const size_t WKE = (size_t)(NKV * HDIM) * HID;      // 1,048,576

  ushort_t* us = (ushort_t*)d_ws;                     // bf16 workspace
  ushort_t* qx = us;             ushort_t* kx = qx + XE;  ushort_t* vx = kx + XE;
  ushort_t* wq = vx + XE;        ushort_t* wk = wq + WQE; ushort_t* wv = wk + WKE;
  ushort_t* wo = wv + WKE;
  ushort_t* Qb = wo + WQE;       ushort_t* Kv = Qb + XE;  ushort_t* Vv = Kv + KVE;
  ushort_t* Cx = Vv + KVE;

  auto cvt = [&](const float* s, ushort_t* d, size_t n) {
    int n4 = (int)(n / 4);
    cvt_f32_bf16<<<(n4 + 255) / 256, 256, 0, stream>>>(s, d, n4);
  };
  cvt(query, qx, XE);  cvt(key, kx, XE);   cvt(value, vx, XE);
  cvt(W_q,   wq, WQE); cvt(W_k, wk, WKE);  cvt(W_v,   wv, WKE);
  cvt(W_o,   wo, WQE);

  gemm_xwt<true><<<dim3(HID / 64, M / 64), 128, 0, stream>>>(qx, wq, Qb, (int)M, HID, HID);
  gemm_xwt<true><<<dim3((NKV * HDIM) / 64, M / 64), 128, 0, stream>>>(kx, wk, Kv, (int)M, NKV * HDIM, HID);
  gemm_xwt<true><<<dim3((NKV * HDIM) / 64, M / 64), 128, 0, stream>>>(vx, wv, Vv, (int)M, NKV * HDIM, HID);

  attn_kernel<<<dim3(SLEN / 16, SB * NHEADS), 32, 0, stream>>>(Qb, Kv, Vv, Cx, attn);

  gemm_xwt<false><<<dim3(HID / 64, M / 64), 128, 0, stream>>>(Cx, wo, out, (int)M, HID, HID);
}
